// CausalSelfAttention_84009560310340
// MI455X (gfx1250) — compile-verified, matrix operand order checked by reading, not run on hardware
//
#include <hip/hip_runtime.h>

#define DEVINL __device__ __forceinline__

typedef __attribute__((ext_vector_type(16))) __bf16 v16bf;
typedef __attribute__((ext_vector_type(16))) unsigned short v16us;
typedef __attribute__((ext_vector_type(8)))  float v8f;

// ---------- bf16 helpers ----------
DEVINL unsigned short f2bfu(float f) {
    unsigned u = __float_as_uint(f);
    unsigned r = u + 0x7FFFu + ((u >> 16) & 1u);   // round-to-nearest-even
    return (unsigned short)(r >> 16);
}
DEVINL __bf16 f2bf(float f) {
    unsigned short s = f2bfu(f);
    __bf16 b;
    __builtin_memcpy(&b, &s, 2);
    return b;
}
DEVINL __bf16 us2bf(unsigned short s) {
    __bf16 b;
    __builtin_memcpy(&b, &s, 2);
    return b;
}

// Scatter 8 bf16 (packed in a uint4) into dst[j*stride], j=0..7.
DEVINL void scatter8(__bf16* dst, int stride, uint4 v) {
    unsigned w[4] = {v.x, v.y, v.z, v.w};
#pragma unroll
    for (int i = 0; i < 4; ++i) {
        dst[(2 * i)     * stride] = us2bf((unsigned short)(w[i] & 0xFFFFu));
        dst[(2 * i + 1) * stride] = us2bf((unsigned short)(w[i] >> 16));
    }
}

// Build a 16x32 bf16 WMMA fragment (A layout; B mirrored with M->N) from a
// row-major LDS tile. ISA 16-bit A 16x32 layout:
//   lane<16 : K in {0..7, 16..23};  lane>=16 : K in {8..15, 24..31}
// Element pairs (2i,2i+1) are consecutive K -> 32-bit LDS reads (fuse to b128).
DEVINL v16bf load_frag(const __bf16* base, int row, int ld, int k0, int half) {
    v16us u;
#pragma unroll
    for (int i = 0; i < 8; ++i) {
        int K = k0 + half * 8 + ((i < 4) ? 0 : 16) + 2 * (i & 3);
        unsigned pr = *(const unsigned*)(base + row * ld + K);
        u[2 * i]     = (unsigned short)(pr & 0xFFFFu);
        u[2 * i + 1] = (unsigned short)(pr >> 16);
    }
    return __builtin_bit_cast(v16bf, u);
}

DEVINL v8f wmma_bf16(v16bf a, v16bf b, v8f c) {
    return __builtin_amdgcn_wmma_f32_16x16x32_bf16(
        /*neg_a=*/false, a, /*neg_b=*/false, b,
        /*c_mod=*/(short)0, c, /*reuse_a=*/false, /*reuse_b=*/false);
}

DEVINL void lds_wait() { asm volatile("s_wait_dscnt 0x0" ::: "memory"); }

// ---------- fp32 -> bf16 conversion ----------
__global__ void cvt_f32_bf16(const float* __restrict__ in, __bf16* __restrict__ out, int n) {
    for (int i = blockIdx.x * blockDim.x + threadIdx.x; i < n; i += gridDim.x * blockDim.x)
        out[i] = f2bf(in[i]);
}

// ---------- GEMM: C[M,N] = A[M,K](bf16) @ B[K,N](bf16) + bias ----------
// 64x64 block tile, 128 threads (4 waves), wave -> 16x64 strip (4 wmma tiles).
template <bool OUT_BF16>
__global__ void gemm64x64(const __bf16* __restrict__ A, const __bf16* __restrict__ Bm,
                          const float* __restrict__ bias, void* __restrict__ Cout,
                          int M, int N, int K) {
    __shared__ __attribute__((aligned(16))) __bf16 As[64 * 32];   // [row][kk]
    __shared__ __attribute__((aligned(16))) __bf16 Bt[64 * 32];   // [n][kk] (K-transposed)
    const int tid  = threadIdx.x;
    const int lane = tid & 31, wave = tid >> 5;
    const int half = lane >> 4, nl = lane & 15;
    const int rowbase = blockIdx.y * 64;
    const int colbase = blockIdx.x * 64;

    v8f acc0 = {}, acc1 = {}, acc2 = {}, acc3 = {};
    for (int kt = 0; kt < K; kt += 32) {
        // A tile: 64x32, 16B-contiguous chunks, coalesced b128 loads/stores.
#pragma unroll
        for (int c = tid; c < 256; c += 128) {
            int row = c >> 2, k8 = (c & 3) * 8;
            *(uint4*)(As + row * 32 + k8) =
                *(const uint4*)(A + (size_t)(rowbase + row) * K + kt + k8);
        }
        // B tile: coalesced b128 global loads of rows, transpose via LDS scatter.
#pragma unroll
        for (int c = tid; c < 256; c += 128) {
            int kk = c >> 3, n8 = (c & 7) * 8;
            uint4 v = *(const uint4*)(Bm + (size_t)(kt + kk) * N + colbase + n8);
            scatter8(Bt + n8 * 32 + kk, 32, v);
        }
        if (kt + 32 < K) {   // prefetch next K-step (global_prefetch_b8)
            __builtin_prefetch(A + (size_t)(rowbase + (tid >> 1)) * K + kt + 32, 0, 1);
            __builtin_prefetch(Bm + (size_t)(kt + 32 + (tid >> 2)) * N + colbase, 0, 1);
        }
        __syncthreads();
        v16bf af = load_frag(As, wave * 16 + nl, 32, 0, half);
        acc0 = wmma_bf16(af, load_frag(Bt, 0 * 16 + nl, 32, 0, half), acc0);
        acc1 = wmma_bf16(af, load_frag(Bt, 1 * 16 + nl, 32, 0, half), acc1);
        acc2 = wmma_bf16(af, load_frag(Bt, 2 * 16 + nl, 32, 0, half), acc2);
        acc3 = wmma_bf16(af, load_frag(Bt, 3 * 16 + nl, 32, 0, half), acc3);
        __syncthreads();
    }
#pragma unroll
    for (int g = 0; g < 4; ++g) {
        int col  = colbase + g * 16 + nl;
        float bv = bias[col];
        v8f a = (g == 0) ? acc0 : (g == 1) ? acc1 : (g == 2) ? acc2 : acc3;
#pragma unroll
        for (int r = 0; r < 8; ++r) {
            int row = rowbase + wave * 16 + r + half * 8;   // C/D layout: lanes16-31 -> M+8
            float val = a[r] + bv;
            if (OUT_BF16) ((__bf16*)Cout)[(size_t)row * N + col] = f2bf(val);
            else          ((float*)Cout)[(size_t)row * N + col]  = val;
        }
    }
}

// ---------- Causal attention ----------
// grid: (B*H, T/64), block: 128 (4 waves). Wave w owns q rows qt*64+w*16..+15.
// Two-pass softmax; writes full fp32 attn_w (zeros above diagonal) and bf16 attn_v.
__global__ void attn_kernel(const __bf16* __restrict__ qkv, float* __restrict__ attnw,
                            __bf16* __restrict__ av) {
    constexpr int T_ = 2048, C_ = 1024, D_ = 64, TC3 = 3072;
    __shared__ __attribute__((aligned(16))) __bf16 Qs[64 * 64];       // [qrow][d]
    __shared__ __attribute__((aligned(16))) __bf16 Ks[4][16 * 64];    // per-wave [kpos][d]
    __shared__ __attribute__((aligned(16))) __bf16 Ps[4][16 * 32];    // per-wave [m][kk]
    __shared__ __attribute__((aligned(16))) __bf16 Vt[4][64 * 32];    // per-wave [d][kk]

    const int bh = blockIdx.x;               // b*H + h
    const int b  = bh >> 4;
    const int h  = bh & 15;
    const int qt = blockIdx.y;
    const int tid = threadIdx.x, lane = tid & 31, wave = tid >> 5;
    const int half = lane >> 4, nl = lane & 15;
    const size_t rowbase_g = (size_t)b * T_ + (size_t)qt * 64;        // row in qkv

    // Q tile: coalesced b128 copy global->LDS
    for (int c = tid; c < 512; c += 128) {
        int r = c >> 3, d8 = (c & 7) * 8;
        *(uint4*)(Qs + r * 64 + d8) =
            *(const uint4*)(qkv + (rowbase_g + r) * TC3 + h * D_ + d8);
    }
    __syncthreads();

    const int qs = qt * 64 + wave * 16;      // first seq index of this wave's strip
    v16bf qa0 = load_frag(Qs, wave * 16 + nl, 64, 0, half);
    v16bf qa1 = load_frag(Qs, wave * 16 + nl, 64, 32, half);
    const float scale = 0.125f;              // 1/sqrt(64)

    float rowM[8], rowL[8];
#pragma unroll
    for (int r = 0; r < 8; ++r) { rowM[r] = -3.0e38f; rowL[r] = 0.0f; }

    __bf16* Kw = Ks[wave];
    __bf16* Pw = Ps[wave];
    __bf16* Vw = Vt[wave];

    // ---- Pass 1: row max / sumexp ----
    const int jlim = (qs + 15) >> 4;
    for (int j = 0; j <= jlim; ++j) {
#pragma unroll
        for (int c = lane; c < 128; c += 32) {
            int kp = c >> 3, d8 = (c & 7) * 8;
            *(uint4*)(Kw + kp * 64 + d8) =
                *(const uint4*)(qkv + ((size_t)b * T_ + j * 16 + kp) * TC3 + C_ + h * D_ + d8);
        }
        lds_wait();
        v8f s = {};
        s = wmma_bf16(qa0, load_frag(Kw, nl, 64, 0, half), s);
        s = wmma_bf16(qa1, load_frag(Kw, nl, 64, 32, half), s);
        const int kg = j * 16 + nl;
#pragma unroll
        for (int r = 0; r < 8; ++r) {
            const int qg = qs + r + half * 8;
            float sv = (kg <= qg) ? s[r] * scale : -3.0e38f;
            float t = sv;
            t = fmaxf(t, __shfl_xor(t, 1));
            t = fmaxf(t, __shfl_xor(t, 2));
            t = fmaxf(t, __shfl_xor(t, 4));
            t = fmaxf(t, __shfl_xor(t, 8));
            float mnew = fmaxf(rowM[r], t);
            float ev = (kg <= qg) ? __expf(sv - mnew) : 0.0f;
            ev += __shfl_xor(ev, 1);
            ev += __shfl_xor(ev, 2);
            ev += __shfl_xor(ev, 4);
            ev += __shfl_xor(ev, 8);
            rowL[r] = rowL[r] * __expf(rowM[r] - mnew) + ev;
            rowM[r] = mnew;
        }
    }
    float rinv[8];
#pragma unroll
    for (int r = 0; r < 8; ++r) rinv[r] = 1.0f / rowL[r];

    // ---- Pass 2: probabilities + O = P @ V ----
    v8f o0 = {}, o1 = {}, o2 = {}, o3 = {};
    const int cmax = (qs + 15) >> 5;
    for (int c = 0; c <= cmax; ++c) {
        const int kt0 = c * 32;
#pragma unroll
        for (int sub = 0; sub < 2; ++sub) {
            const int j = 2 * c + sub;
#pragma unroll
            for (int cc = lane; cc < 128; cc += 32) {
                int kp = cc >> 3, d8 = (cc & 7) * 8;
                *(uint4*)(Kw + kp * 64 + d8) =
                    *(const uint4*)(qkv + ((size_t)b * T_ + j * 16 + kp) * TC3 + C_ + h * D_ + d8);
            }
            lds_wait();
            v8f s = {};
            s = wmma_bf16(qa0, load_frag(Kw, nl, 64, 0, half), s);
            s = wmma_bf16(qa1, load_frag(Kw, nl, 64, 32, half), s);
            const int kg = j * 16 + nl;
#pragma unroll
            for (int r = 0; r < 8; ++r) {
                const int qg = qs + r + half * 8;
                float p = (kg <= qg) ? __expf(s[r] * scale - rowM[r]) * rinv[r] : 0.0f;
                const int m = r + half * 8;
                attnw[((size_t)bh * T_ + qs + m) * T_ + kg] = p;
                Pw[m * 32 + sub * 16 + nl] = f2bf(p);
            }
        }
        // V chunk: coalesced b128 row loads, transposed into LDS: Vw[d*32+kk]=v[kt0+kk][d]
#pragma unroll
        for (int cc = lane; cc < 256; cc += 32) {
            int kk = cc >> 3, d8 = (cc & 7) * 8;
            uint4 v = *(const uint4*)(qkv + ((size_t)b * T_ + kt0 + kk) * TC3 + 2 * C_ + h * D_ + d8);
            scatter8(Vw + d8 * 32 + kk, 32, v);
        }
        lds_wait();
        v16bf pa = load_frag(Pw, nl, 32, 0, half);
        o0 = wmma_bf16(pa, load_frag(Vw, 0 * 16 + nl, 32, 0, half), o0);
        o1 = wmma_bf16(pa, load_frag(Vw, 1 * 16 + nl, 32, 0, half), o1);
        o2 = wmma_bf16(pa, load_frag(Vw, 2 * 16 + nl, 32, 0, half), o2);
        o3 = wmma_bf16(pa, load_frag(Vw, 3 * 16 + nl, 32, 0, half), o3);
    }
    // zero-fill fully masked attn_w region
    for (int c = cmax + 1; c < T_ / 32; ++c) {
#pragma unroll
        for (int r = 0; r < 8; ++r) {
            const int m = r + half * 8;
            size_t base = ((size_t)bh * T_ + qs + m) * (size_t)T_ + c * 32;
            attnw[base + nl]      = 0.0f;
            attnw[base + 16 + nl] = 0.0f;
        }
    }
    // write attn_v (bf16) for the output projection
#pragma unroll
    for (int g = 0; g < 4; ++g) {
        v8f a = (g == 0) ? o0 : (g == 1) ? o1 : (g == 2) ? o2 : o3;
#pragma unroll
        for (int r = 0; r < 8; ++r) {
            const int m = r + half * 8;
            av[(rowbase_g + wave * 16 + m) * C_ + h * D_ + g * 16 + nl] = f2bf(a[r]);
        }
    }
}

extern "C" void kernel_launch(void* const* d_in, const int* in_sizes, int n_in,
                              void* d_out, int out_size, void* d_ws, size_t ws_size,
                              hipStream_t stream) {
    const float* x     = (const float*)d_in[0];
    const float* w_qkv = (const float*)d_in[1];
    const float* b_qkv = (const float*)d_in[2];
    const float* w_o   = (const float*)d_in[3];
    const float* b_o   = (const float*)d_in[4];

    constexpr int Bc = 2, Tc = 2048, Cc = 1024, Hc = 16;
    constexpr int BT = Bc * Tc;     // 4096
    constexpr int C3 = 3 * Cc;      // 3072

    __bf16* xb    = (__bf16*)d_ws;
    __bf16* wqkvb = xb    + (size_t)BT * Cc;
    __bf16* wob   = wqkvb + (size_t)Cc * C3;
    __bf16* qkvb  = wob   + (size_t)Cc * Cc;
    __bf16* avb   = qkvb  + (size_t)BT * C3;

    float* o_out = (float*)d_out;
    float* attnw = o_out + (size_t)BT * Cc;

    cvt_f32_bf16<<<1024, 256, 0, stream>>>(x, xb, BT * Cc);
    cvt_f32_bf16<<<1024, 256, 0, stream>>>(w_qkv, wqkvb, Cc * C3);
    cvt_f32_bf16<<<1024, 256, 0, stream>>>(w_o, wob, Cc * Cc);

    gemm64x64<true ><<<dim3(C3 / 64, BT / 64), 128, 0, stream>>>(xb, wqkvb, b_qkv, qkvb, BT, C3, Cc);
    attn_kernel<<<dim3(Bc * Hc, Tc / 64), 128, 0, stream>>>(qkvb, attnw, avb);
    gemm64x64<false><<<dim3(Cc / 64, BT / 64), 128, 0, stream>>>(avb, wob, b_o, o_out, BT, Cc, Cc);

    (void)in_sizes; (void)n_in; (void)out_size; (void)ws_size;
}
